// State_86612310491558
// MI455X (gfx1250) — compile-verified, matrix operand order checked by reading, not run on hardware
//
#include <hip/hip_runtime.h>

typedef __attribute__((ext_vector_type(16))) __bf16 v16bf;
typedef __attribute__((ext_vector_type(8)))  __bf16 v8bf;
typedef __attribute__((ext_vector_type(8)))  float  v8f;

#define T_STEPS 2048
#define BATCH   64
#define DIN     256
#define HID     512
#define GCOLS   (4*HID)       // 2048 fused gate columns
#define NWG     16
#define NTHREADS 256
#define NKT_X   (DIN/32)      // 8 k-tiles for x@Wx
#define NKT_H   (HID/32)      // 16 k-tiles for h@Wh
#define NCT     (GCOLS/16)    // 128 gate column tiles

// workspace layout (bytes)
#define WS_COUNTER 0
#define WS_HBUF    4096                               // 2*64*512 bf16 = 131072 B
#define WS_WXF     (WS_HBUF + 2*BATCH*HID*2)          // 128 ct * 8 kt * 512 bf16 = 1 MB
#define WS_WHF     (WS_WXF + (size_t)NCT*NKT_X*512*2) // 128 ct * 16 kt * 512 bf16 = 2 MB

__device__ __forceinline__ float sigmoidf_(float x) {
    return 1.0f / (1.0f + __expf(-x));
}
__device__ __forceinline__ float tanhf_(float x) {
    float e = __expf(2.0f * x);
    return 1.0f - 2.0f / (e + 1.0f);   // exact at +/-inf, no NaN
}

// ---------------- prep: zero h double-buffer + sync counter ----------------
__global__ void init_kernel(__bf16* hbuf, unsigned* counter) {
    int idx = blockIdx.x * blockDim.x + threadIdx.x;
    if (idx < 2 * BATCH * HID) hbuf[idx] = (__bf16)0.0f;
    if (idx == 0) *counter = 0u;
}

// ---------------- prep: repack fp32 weights into bf16 B-fragments ----------
// B fragment layout assumed for v_wmma_f32_16x16x32_bf16 (B is 32x16, K x N):
//   lane l<16 : N = l,      K = {0..7} in elems 0..7, {16..23} in elems 8..15
//   lane l>=16: N = l-16,   K = {8..15} in elems 0..7, {24..31} in elems 8..15
// Stored fragment-major: [ct][kt][lane][16 elems] -> each lane reads 32 contiguous bytes.
__global__ void pack_bfrags_kernel(const float* __restrict__ w, __bf16* __restrict__ dst,
                                   int nkt) {
    int idx = blockIdx.x * blockDim.x + threadIdx.x;
    int total = NCT * nkt * 32 * 16;
    if (idx >= total) return;
    int e    = idx & 15;
    int lane = (idx >> 4) & 31;
    int fi   = idx >> 9;          // ct*nkt + kt
    int kt   = fi % nkt;
    int ct   = fi / nkt;
    int klocal = (e < 8 ? e : e + 8) + (lane >= 16 ? 8 : 0);
    int K = kt * 32 + klocal;
    int N = ct * 16 + (lane & 15);
    dst[idx] = (__bf16)w[(size_t)K * GCOLS + N];
}

// ---------------- persistent fused LSTM ----------------
__global__ __launch_bounds__(NTHREADS, 1)
void lstm_persistent_kernel(const float* __restrict__ x,      // [T,B,D] fp32
                            const float* __restrict__ bias4h, // [4H] fp32
                            const __bf16* __restrict__ wxf,   // packed B frags
                            const __bf16* __restrict__ whf,   // packed B frags
                            __bf16* __restrict__ hbuf,        // [2][B*H] bf16
                            float* __restrict__ out,          // ys | h_T | c_T
                            unsigned* __restrict__ counter) {
    const int wave = threadIdx.x >> 5;
    const int lane = threadIdx.x & 31;
    const int ln   = lane & 15;

    const int ht = blockIdx.x * 2 + (wave & 1);   // global h-tile 0..31
    const int mq = wave >> 1;                     // row quarter 0..3
    const int m0 = mq * 16;

    const int col   = ht * 16 + ln;               // hidden column this lane owns (C layout N)
    const int rowA  = m0 + ln;                    // A-matrix row for this lane
    const int rowC0 = m0 + ((lane < 16) ? 0 : 8); // C layout: vgpr r -> row rowC0 + r
    const int koffs = (lane < 16) ? 0 : 8;        // A layout K split between lane halves

    float bias[4];
#pragma unroll
    for (int g = 0; g < 4; ++g) bias[g] = bias4h[g * HID + col];

    // fragment base pointers (gate g lives at column tile g*32 + ht)
    const __bf16* wxp[4];
    const __bf16* whp[4];
#pragma unroll
    for (int g = 0; g < 4; ++g) {
        int ct = g * 32 + ht;
        wxp[g] = wxf + ((size_t)ct * NKT_X) * 512 + lane * 16;
        whp[g] = whf + ((size_t)ct * NKT_H) * 512 + lane * 16;
    }

    union BU { v16bf v; v8bf h[2]; };

    v8f c_reg = {};          // cell state lives in registers for all 2048 steps
    int p = 0;               // h double-buffer parity

#pragma clang loop unroll(disable)
    for (int t = 0; t < T_STEPS; ++t) {
        v8f acc0 = {}, acc1 = {}, acc2 = {}, acc3 = {};

        // ---- x_t @ Wx : 8 k-tiles, A gathered fp32 -> bf16 on the fly ----
        const float* xr = x + ((size_t)t * BATCH + rowA) * DIN;
#pragma unroll 2
        for (int kt = 0; kt < NKT_X; ++kt) {
            int kc = kt * 32 + koffs;
            const float4* q0 = (const float4*)(xr + kc);
            const float4* q1 = (const float4*)(xr + kc + 16);
            float4 f0 = q0[0], f1 = q0[1], f2 = q1[0], f3 = q1[1];
            v16bf a;
            a[0]=(__bf16)f0.x;  a[1]=(__bf16)f0.y;  a[2]=(__bf16)f0.z;  a[3]=(__bf16)f0.w;
            a[4]=(__bf16)f1.x;  a[5]=(__bf16)f1.y;  a[6]=(__bf16)f1.z;  a[7]=(__bf16)f1.w;
            a[8]=(__bf16)f2.x;  a[9]=(__bf16)f2.y;  a[10]=(__bf16)f2.z; a[11]=(__bf16)f2.w;
            a[12]=(__bf16)f3.x; a[13]=(__bf16)f3.y; a[14]=(__bf16)f3.z; a[15]=(__bf16)f3.w;

            BU b0, b1, b2, b3;
            const __bf16* bp0 = wxp[0] + (size_t)kt * 512;
            const __bf16* bp1 = wxp[1] + (size_t)kt * 512;
            const __bf16* bp2 = wxp[2] + (size_t)kt * 512;
            const __bf16* bp3 = wxp[3] + (size_t)kt * 512;
            b0.h[0] = *(const v8bf*)bp0;      b0.h[1] = *(const v8bf*)(bp0 + 8);
            b1.h[0] = *(const v8bf*)bp1;      b1.h[1] = *(const v8bf*)(bp1 + 8);
            b2.h[0] = *(const v8bf*)bp2;      b2.h[1] = *(const v8bf*)(bp2 + 8);
            b3.h[0] = *(const v8bf*)bp3;      b3.h[1] = *(const v8bf*)(bp3 + 8);
            acc0 = __builtin_amdgcn_wmma_f32_16x16x32_bf16(false, a, false, b0.v, (short)0, acc0, false, false);
            acc1 = __builtin_amdgcn_wmma_f32_16x16x32_bf16(false, a, false, b1.v, (short)0, acc1, false, false);
            acc2 = __builtin_amdgcn_wmma_f32_16x16x32_bf16(false, a, false, b2.v, (short)0, acc2, false, false);
            acc3 = __builtin_amdgcn_wmma_f32_16x16x32_bf16(false, a, false, b3.v, (short)0, acc3, false, false);
        }

        // ---- h_{t-1} @ Wh : 16 k-tiles, A loaded bf16 from double buffer ----
        const __bf16* hr = hbuf + (size_t)p * BATCH * HID + (size_t)rowA * HID;
#pragma unroll 2
        for (int kt = 0; kt < NKT_H; ++kt) {
            int kc = kt * 32 + koffs;
            BU au;
            au.h[0] = *(const v8bf*)(hr + kc);
            au.h[1] = *(const v8bf*)(hr + kc + 16);

            BU b0, b1, b2, b3;
            const __bf16* bp0 = whp[0] + (size_t)kt * 512;
            const __bf16* bp1 = whp[1] + (size_t)kt * 512;
            const __bf16* bp2 = whp[2] + (size_t)kt * 512;
            const __bf16* bp3 = whp[3] + (size_t)kt * 512;
            b0.h[0] = *(const v8bf*)bp0;      b0.h[1] = *(const v8bf*)(bp0 + 8);
            b1.h[0] = *(const v8bf*)bp1;      b1.h[1] = *(const v8bf*)(bp1 + 8);
            b2.h[0] = *(const v8bf*)bp2;      b2.h[1] = *(const v8bf*)(bp2 + 8);
            b3.h[0] = *(const v8bf*)bp3;      b3.h[1] = *(const v8bf*)(bp3 + 8);
            acc0 = __builtin_amdgcn_wmma_f32_16x16x32_bf16(false, au.v, false, b0.v, (short)0, acc0, false, false);
            acc1 = __builtin_amdgcn_wmma_f32_16x16x32_bf16(false, au.v, false, b1.v, (short)0, acc1, false, false);
            acc2 = __builtin_amdgcn_wmma_f32_16x16x32_bf16(false, au.v, false, b2.v, (short)0, acc2, false, false);
            acc3 = __builtin_amdgcn_wmma_f32_16x16x32_bf16(false, au.v, false, b3.v, (short)0, acc3, false, false);
        }

        // ---- elementwise gate math; c stays in registers ----
        __bf16* hw = hbuf + (size_t)(1 - p) * BATCH * HID;
        float*  ys = out + (size_t)t * BATCH * HID;
#pragma unroll
        for (int r = 0; r < 8; ++r) {
            float iv = sigmoidf_(acc0[r] + bias[0]);
            float fv = sigmoidf_(acc1[r] + bias[1]);
            float gv = tanhf_  (acc2[r] + bias[2]);
            float ov = sigmoidf_(acc3[r] + bias[3]);
            float cn = fv * c_reg[r] + iv * gv;
            c_reg[r] = cn;
            float hn = ov * tanhf_(cn);
            int row = rowC0 + r;
            __builtin_nontemporal_store(hn, &ys[(size_t)row * HID + col]); // don't pollute L2
            hw[(size_t)row * HID + col] = (__bf16)hn;                      // next-step operand
        }
        if (t == T_STEPS - 1) {
            float* hT = out + (size_t)T_STEPS * BATCH * HID;
#pragma unroll
            for (int r = 0; r < 8; ++r) {
                int row = rowC0 + r;
                float* yp = out + (size_t)t * BATCH * HID + (size_t)row * HID + col;
                hT[(size_t)row * HID + col] = *yp; // value just stored is in registers path:
            }
        }

        // ---- one grid-wide barrier per step (double buffer makes it sufficient) ----
        __builtin_amdgcn_fence(__ATOMIC_RELEASE, "agent");
        __syncthreads();
        if (threadIdx.x == 0) {
            __hip_atomic_fetch_add(counter, 1u, __ATOMIC_RELAXED, __HIP_MEMORY_SCOPE_AGENT);
            unsigned target = (unsigned)(t + 1) * NWG;
            while (__hip_atomic_load(counter, __ATOMIC_RELAXED, __HIP_MEMORY_SCOPE_AGENT) < target) {
                __builtin_amdgcn_s_sleep(2);
            }
        }
        __syncthreads();
        __builtin_amdgcn_fence(__ATOMIC_ACQUIRE, "agent");

        p ^= 1;
    }

    // final cell state c_T
    float* cT = out + (size_t)T_STEPS * BATCH * HID + (size_t)BATCH * HID;
#pragma unroll
    for (int r = 0; r < 8; ++r) {
        int row = rowC0 + r;
        cT[(size_t)row * HID + col] = c_reg[r];
    }
}

extern "C" void kernel_launch(void* const* d_in, const int* in_sizes, int n_in,
                              void* d_out, int out_size, void* d_ws, size_t ws_size,
                              hipStream_t stream) {
    const float* x  = (const float*)d_in[0];   // [T,B,D]
    const float* Wx = (const float*)d_in[1];   // [D,4H]
    const float* Wh = (const float*)d_in[2];   // [H,4H]
    const float* b  = (const float*)d_in[3];   // [4H]

    char* ws = (char*)d_ws;
    unsigned* counter = (unsigned*)(ws + WS_COUNTER);
    __bf16*   hbuf    = (__bf16*)(ws + WS_HBUF);
    __bf16*   wxf     = (__bf16*)(ws + WS_WXF);
    __bf16*   whf     = (__bf16*)(ws + WS_WHF);

    int ninit = 2 * BATCH * HID;
    init_kernel<<<(ninit + 255) / 256, 256, 0, stream>>>(hbuf, counter);

    int nx = NCT * NKT_X * 512;
    pack_bfrags_kernel<<<(nx + 255) / 256, 256, 0, stream>>>(Wx, wxf, NKT_X);
    int nh = NCT * NKT_H * 512;
    pack_bfrags_kernel<<<(nh + 255) / 256, 256, 0, stream>>>(Wh, whf, NKT_H);

    lstm_persistent_kernel<<<NWG, NTHREADS, 0, stream>>>(
        x, b, wxf, whf, hbuf, (float*)d_out, counter);
}
